// TransformerEncoder_21036749815922
// MI455X (gfx1250) — compile-verified, matrix operand order checked by reading, not run on hardware
//
#include <hip/hip_runtime.h>
#include <hip/hip_bf16.h>
#include <math.h>

// ---------------- problem constants ----------------
#define BB   4
#define LL   512
#define HH   1024
#define NHH  16
#define DKK  64
#define DFFv 4096
#define NLv  4
#define NEGV (-1e18f)
#define INV_SQRT2 0.70710678118654752f

typedef __attribute__((ext_vector_type(16))) __bf16   v16bf;
typedef __attribute__((ext_vector_type(16))) _Float16 v16h;
typedef __attribute__((ext_vector_type(8)))  float    v8f;

// WMMA 16-bit A/B fragment: lane half hlf, element e -> K = ((e<8)?e:e+8) + hlf*8.
// Inverse: k -> hlf=(k>>3)&1, e=(k&7)+((k&16)?8:0). 8 aligned consecutive k map to 8
// consecutive e with constant hlf -> swizzled stores are contiguous 16B.

__device__ __forceinline__ v8f wmma_bf16(v16bf a, v16bf b, v8f c) {
    return __builtin_amdgcn_wmma_f32_16x16x32_bf16(false, a, false, b, (short)0, c, false, false);
}
__device__ __forceinline__ v8f wmma_f16(v16h a, v16h b, v8f c) {
    return __builtin_amdgcn_wmma_f32_16x16x32_f16(false, a, false, b, (short)0, c, false, false);
}

// LDS byte offset of a shared-memory pointer (for async-to-LDS VDST operand).
__device__ __forceinline__ unsigned lds_offset(const void* p) {
    return (unsigned)(unsigned long long)(__attribute__((address_space(3))) void*)p;
}

// ---------------- layernorm (row staged via async global->LDS copy) ----------------
__global__ __launch_bounds__(256) void ln_kernel(const float* __restrict__ x,
                                                 const float* __restrict__ g,
                                                 const float* __restrict__ bta,
                                                 float* __restrict__ out) {
    __shared__ __align__(16) float xs[HH];
    __shared__ float r1[256];
    __shared__ float r2[256];
    const int row = blockIdx.x;
    const int tid = threadIdx.x;
    const float* xr = x + (size_t)row * HH;
    // 256 threads x 16B async copy = 4KB row (tracked by ASYNCcnt)
    {
        unsigned loff = lds_offset(&xs[tid * 4]);
        const float* gp = xr + tid * 4;
        asm volatile("global_load_async_to_lds_b128 %0, %1, off"
                     :: "v"(loff), "v"(gp) : "memory");
        asm volatile("s_wait_asynccnt 0x0" ::: "memory");
    }
    __syncthreads();
    float s = 0.f, s2 = 0.f;
    for (int j = tid; j < HH; j += 256) { float v = xs[j]; s += v; s2 += v * v; }
    r1[tid] = s; r2[tid] = s2;
    __syncthreads();
    for (int o = 128; o > 0; o >>= 1) {
        if (tid < o) { r1[tid] += r1[tid + o]; r2[tid] += r2[tid + o]; }
        __syncthreads();
    }
    const float mu  = r1[0] * (1.0f / HH);
    const float var = r2[0] * (1.0f / HH) - mu * mu;
    const float rstd = rsqrtf(var + 1e-5f);
    float* orow = out + (size_t)row * HH;
    for (int j = tid; j < HH; j += 256)
        orow[j] = (xs[j] - mu) * rstd * g[j] + bta[j];
}

// ---------------- rotary sin/cos table ----------------
__global__ __launch_bounds__(256) void sincos_kernel(const int* __restrict__ leaf,
                                                     float* __restrict__ sinb,
                                                     float* __restrict__ cosb) {
    int t = blockIdx.x * blockDim.x + threadIdx.x;
    if (t >= BB * LL * (DKK / 2)) return;
    int i  = t & 31;
    int bl = t >> 5;
    float pos = (float)leaf[bl];
    float inv = __expf(-((2.0f * (float)i) / (float)DKK) * logf(10000.0f));
    float ang = pos * inv;
    sinb[t] = sinf(ang);
    cosb[t] = cosf(ang);
}

// ---------------- tiny gate projection: out[row,h] = sigmoid(h_row . Wg[:,h] + bg[h]) ----
__global__ __launch_bounds__(128) void gate_kernel(const float* __restrict__ hbuf,
                                                   const float* __restrict__ Wg,
                                                   const float* __restrict__ bg,
                                                   float* __restrict__ out) {
    __shared__ float hs[HH];
    const int row = blockIdx.x;
    const int tid = threadIdx.x;
    const float* hr = hbuf + (size_t)row * HH;
    for (int j = tid; j < HH; j += 128) hs[j] = hr[j];
    __syncthreads();
    const int head = tid >> 3, part = tid & 7;
    float s = 0.f;
    for (int j = part; j < HH; j += 8) s += hs[j] * Wg[(size_t)j * NHH + head];
    s += __shfl_xor(s, 1, 32);
    s += __shfl_xor(s, 2, 32);
    s += __shfl_xor(s, 4, 32);
    if (part == 0)
        out[(size_t)row * NHH + head] = 1.0f / (1.0f + __expf(-(s + bg[head])));
}

// ---------------- pack Q/K into bf16 WMMA fragment layout (plain + rotary) ----------------
// out layout: [bh][tile(32)][kk(2)][lane(32)][e(16)] bf16; frag = 32B contiguous per lane.
__global__ __launch_bounds__(256) void pack_qk_kernel(const float* __restrict__ src,
                                                      const float* __restrict__ sinb,
                                                      const float* __restrict__ cosb,
                                                      __bf16* __restrict__ outp,
                                                      __bf16* __restrict__ outr) {
    const int t  = blockIdx.x * 256 + threadIdx.x;   // B*NH*32*2*32*2 = 262144 threads
    const int eb = t & 1;
    const int l  = (t >> 1) & 31;
    const int kk = (t >> 6) & 1;
    const int qt = (t >> 7) & 31;
    const int bh = t >> 12;
    const int b  = bh >> 4;
    const int hh = bh & 15;
    const int r   = l & 15, hlf = l >> 4;
    const int pos = qt * 16 + r;
    const int dbase = kk * 32 + eb * 16 + hlf * 8;
    const size_t srow = ((size_t)b * LL + pos) * HH + hh * DKK + dbase;
    const size_t orow = ((size_t)((bh * 32 + qt) * 2 + kk) * 32 + l) * 16 + eb * 8;
    float x[8];
#pragma unroll
    for (int j = 0; j < 8; j++) x[j] = src[srow + j];
#pragma unroll
    for (int j = 0; j < 8; j++) outp[orow + j] = (__bf16)x[j];
    const size_t sc = ((size_t)b * LL + pos) * 32 + (dbase >> 1);
#pragma unroll
    for (int j = 0; j < 4; j++) {
        float sn = sinb[sc + j], cs = cosb[sc + j];
        float t0 = x[2 * j] * cs - x[2 * j + 1] * sn;
        float t1 = x[2 * j + 1] * cs + x[2 * j] * sn;
        outr[orow + 2 * j]     = (__bf16)t0;
        outr[orow + 2 * j + 1] = (__bf16)t1;
    }
}

// ---------------- pack V into f16 B-fragment layout for ctx GEMM ----------------
// out layout: [bh][nt(4)][kk(16)][lane(32)][e(16)] f16. B[k=key][n=d] fragment.
__global__ __launch_bounds__(256) void pack_v_kernel(const float* __restrict__ src,
                                                     _Float16* __restrict__ outv) {
    const int t  = blockIdx.x * 256 + threadIdx.x;   // 64*4*16*32*2 = 262144
    const int eb = t & 1;
    const int l  = (t >> 1) & 31;
    const int kk = (t >> 6) & 15;
    const int nt = (t >> 10) & 3;
    const int bh = t >> 12;
    const int b  = bh >> 4;
    const int hh = bh & 15;
    const int nl = l & 15, hlf = l >> 4;
    const int d    = nt * 16 + nl;
    const int keyb = kk * 32 + eb * 16 + hlf * 8;
    const size_t obase = ((size_t)(((bh * 4 + nt) * 16 + kk) * 32) + l) * 16 + eb * 8;
#pragma unroll
    for (int j = 0; j < 8; j++)
        outv[obase + j] = (_Float16)src[((size_t)b * LL + keyb + j) * HH + hh * DKK + d];
}

// ---------------- general GEMM: D = act(alpha*(A@B + bias)) (+ res) ----------------
// A:[M,K] fp32, B:[K,N] fp32 row-major. REQUIRES M%128==0, N%64==0, K%32==0.
// 128x64 block tile, 8 waves, each wave computes 32x32 (4 WMMA / K-step).
// Software pipeline: issue global loads (t+1) -> WMMA (t) -> cvt+store LDS (t+1).
__global__ __launch_bounds__(256, 2) void gemm_kernel(const float* __restrict__ A,
                                                      const float* __restrict__ Bm,
                                                      const float* __restrict__ bias,
                                                      const float* __restrict__ res,
                                                      float* __restrict__ D,
                                                      int M, int N, int K,
                                                      float alpha, int act) {
    __shared__ uint As[2][8][32][8];   // [buf][mi][lane][8 uints = 16 bf16] = 16KB
    __shared__ uint Bs[2][4][32][8];   // 8KB
    const int tid  = threadIdx.x;
    const int w    = tid >> 5;
    const int lane = tid & 31;
    const int hlf  = lane >> 4;
    const int l15  = lane & 15;
    const int m0 = blockIdx.y * 128, n0 = blockIdx.x * 64;
    const int mi0 = (w & 3) * 2, mi1 = mi0 + 1;
    const int nj0 = (w >> 2) * 2, nj1 = nj0 + 1;

    // A loader: thread -> (row ar 0..127, 16 consecutive k at acb in {0,16})
    const int ar = tid >> 1, acb = (tid & 1) * 16;
    const int a_mi = ar >> 4, arr = ar & 15;
    const int a_eb = (acb & 16) ? 4 : 0;     // uint e-offset, same for both k-halves
    // B loader: thread -> (col bn 0..63, k-group bkg of 8 rows)
    const int bn = tid & 63, bkg = tid >> 6;
    const int b_nt = bn >> 4;
    const int b_ln = ((bkg & 1) << 4) | (bn & 15);
    const int b_eb = (bkg & 2) ? 4 : 0;
    const int nk = K >> 5;

    float ra[16], rb[8];
    auto load_regs = [&](int t) {
        const float* Ap = A + (size_t)(m0 + ar) * K + t * 32 + acb;
#pragma unroll
        for (int j = 0; j < 16; j++) ra[j] = Ap[j];
        const float* Bp = Bm + (size_t)(t * 32 + bkg * 8) * N + n0 + bn;
#pragma unroll
        for (int j = 0; j < 8; j++) rb[j] = Bp[(size_t)j * N];
        if (t + 1 < nk) {
            __builtin_prefetch(Ap + 32, 0, 1);
            __builtin_prefetch(Bp + (size_t)32 * N, 0, 1);
        }
    };
    auto store_lds = [&](int buf) {
        union { __bf16 bh[8]; uint u[4]; } t0, t1, tb;
#pragma unroll
        for (int j = 0; j < 8; j++) { t0.bh[j] = (__bf16)ra[j]; t1.bh[j] = (__bf16)ra[8 + j]; }
#pragma unroll
        for (int j = 0; j < 4; j++) As[buf][a_mi][arr][a_eb + j]      = t0.u[j];
#pragma unroll
        for (int j = 0; j < 4; j++) As[buf][a_mi][16 + arr][a_eb + j] = t1.u[j];
#pragma unroll
        for (int j = 0; j < 8; j++) tb.bh[j] = (__bf16)rb[j];
#pragma unroll
        for (int j = 0; j < 4; j++) Bs[buf][b_nt][b_ln][b_eb + j] = tb.u[j];
    };

    v8f c00 = {}, c01 = {}, c10 = {}, c11 = {};
    load_regs(0);
    store_lds(0);
    __syncthreads();
    for (int t = 0; t < nk; t++) {
        const int cur = t & 1;
        if (t + 1 < nk) load_regs(t + 1);
        union { uint u[8]; v16bf v; } fa0, fa1, fb0, fb1;
#pragma unroll
        for (int i = 0; i < 8; i++) {
            fa0.u[i] = As[cur][mi0][lane][i];
            fa1.u[i] = As[cur][mi1][lane][i];
            fb0.u[i] = Bs[cur][nj0][lane][i];
            fb1.u[i] = Bs[cur][nj1][lane][i];
        }
        c00 = wmma_bf16(fa0.v, fb0.v, c00);
        c01 = wmma_bf16(fa0.v, fb1.v, c01);
        c10 = wmma_bf16(fa1.v, fb0.v, c10);
        c11 = wmma_bf16(fa1.v, fb1.v, c11);
        if (t + 1 < nk) store_lds(cur ^ 1);
        __syncthreads();
    }

    auto epi = [&](const v8f& c, int mi, int nj) {
#pragma unroll
        for (int r = 0; r < 8; r++) {
            const int gm = m0 + mi * 16 + r + hlf * 8;
            const int gn = n0 + nj * 16 + l15;
            float v = alpha * (c[r] + (bias ? bias[gn] : 0.0f));
            if (act == 1)      v = 0.5f * v * (1.0f + erff(v * INV_SQRT2));
            else if (act == 2) v = 1.0f / (1.0f + __expf(-v));
            if (res) v += res[(size_t)gm * N + gn];
            D[(size_t)gm * N + gn] = v;
        }
    };
    epi(c00, mi0, nj0);
    epi(c01, mi0, nj1);
    epi(c10, mi1, nj0);
    epi(c11, mi1, nj1);
}

// ---------------- fused dual attention ----------------
// grid (L/16, NH, B), 256 threads. Fragments are pre-packed -> coalesced b128 loads.
__global__ __launch_bounds__(256, 1) void attn_kernel(const __bf16* __restrict__ qpk,
                                                      const __bf16* __restrict__ qrt,
                                                      const __bf16* __restrict__ kpk,
                                                      const __bf16* __restrict__ krt,
                                                      const _Float16* __restrict__ vpk,
                                                      const float* __restrict__ gateb,
                                                      const float* __restrict__ tree_score,
                                                      const int* __restrict__ leaf,
                                                      float* __restrict__ ctx) {
    __shared__ float    S[16][LL];         // 32 KB
    __shared__ _Float16 Psw[16 * 32 * 16]; // 16 KB, swizzled [kk][lane][e]
    const int tid  = threadIdx.x;
    const int w    = tid >> 5;
    const int lane = tid & 31;
    const int hlf  = lane >> 4;
    const int l15  = lane & 15;
    const int qt = blockIdx.x;
    const int hh = blockIdx.y;
    const int b  = blockIdx.z;
    const int bh = b * NHH + hh;
    const int q0 = qt * 16;
    const size_t bL = (size_t)b * LL;

    for (int pass = 0; pass < 2; ++pass) {
        const v16bf* qv = (const v16bf*)(pass ? qrt : qpk);
        const v16bf* kv = (const v16bf*)(pass ? krt : kpk);
        const v16bf a0 = qv[((size_t)(bh * 32 + qt) * 2 + 0) * 32 + lane];
        const v16bf a1 = qv[((size_t)(bh * 32 + qt) * 2 + 1) * 32 + lane];
#pragma unroll 1
        for (int t = 0; t < 4; t++) {
            const int nt = w + t * 8;
            const v16bf b0 = kv[((size_t)(bh * 32 + nt) * 2 + 0) * 32 + lane];
            const v16bf b1 = kv[((size_t)(bh * 32 + nt) * 2 + 1) * 32 + lane];
            v8f acc = {};
            acc = wmma_bf16(a0, b0, acc);
            acc = wmma_bf16(a1, b1, acc);
#pragma unroll
            for (int r = 0; r < 8; r++) S[r + hlf * 8][nt * 16 + l15] = acc[r];
        }
        __syncthreads();

        // masking + softmax: wave w owns rows 2w, 2w+1
        for (int rr = 0; rr < 2; rr++) {
            const int row = 2 * w + rr;
            const int qp  = q0 + row;
            const bool qvalid = leaf[bL + qp] != 0;
            float mx = -3.0e38f;
            for (int j = lane; j < LL; j += 32) {
                float v = S[row][j];
                if (pass == 0) {
                    v = (leaf[bL + j] == 0)
                            ? NEGV
                            : (v + tree_score[(((size_t)b * NHH + hh) * LL + qp) * LL + j]) * INV_SQRT2;
                } else {
                    v = (qvalid && leaf[bL + j] != 0) ? v : NEGV;
                }
                S[row][j] = v;
                mx = fmaxf(mx, v);
            }
#pragma unroll
            for (int o = 16; o > 0; o >>= 1) mx = fmaxf(mx, __shfl_xor(mx, o, 32));
            float sum = 0.f;
            for (int j = lane; j < LL; j += 32) {
                float e = __expf(S[row][j] - mx);
                S[row][j] = e;
                sum += e;
            }
#pragma unroll
            for (int o = 16; o > 0; o >>= 1) sum += __shfl_xor(sum, o, 32);
            const float inv = 1.0f / sum;
            const float g  = (leaf[bL + qp] == 0) ? 1.0f : gateb[(bL + qp) * NHH + hh];
            const float gf = pass ? (1.0f - g) : g;
            for (int j = lane; j < LL; j += 32) {
                const float pv = S[row][j] * inv * gf;
                const int kkb = j >> 5, jin = j & 31;
                const int hj  = (jin >> 3) & 1;
                const int e   = (jin & 7) + ((jin & 16) ? 8 : 0);
                const int idx = ((kkb * 32) + hj * 16 + row) * 16 + e;
                if (pass == 0) Psw[idx] = (_Float16)pv;
                else           Psw[idx] = (_Float16)((float)Psw[idx] + pv);
            }
        }
        __syncthreads();
    }

    // ctx = P(16x512 f16) @ V(512x64): waves 0..3, one 16-wide d-subtile each
    if (w < 4) {
        const v16h* pv = (const v16h*)Psw;
        const v16h* vv = (const v16h*)vpk;
        const int dn = w * 16 + l15;
        v8f acc = {};
#pragma unroll 1
        for (int kk = 0; kk < 16; kk++) {
            const v16h af = pv[kk * 32 + lane];
            const v16h bf = vv[((size_t)(bh * 4 + w) * 16 + kk) * 32 + lane];
            acc = wmma_f16(af, bf, acc);
        }
#pragma unroll
        for (int r = 0; r < 8; r++)
            ctx[(bL + q0 + r + hlf * 8) * HH + hh * DKK + dn] = acc[r];
    }
}

// ---------------- host driver ----------------
extern "C" void kernel_launch(void* const* d_in, const int* in_sizes, int n_in,
                              void* d_out, int out_size, void* d_ws, size_t ws_size,
                              hipStream_t stream) {
    const float* content    = (const float*)d_in[0];
    const float* tree_score = (const float*)d_in[1];
    const float* Wq = (const float*)d_in[2];
    const float* bq = (const float*)d_in[3];
    const float* Wk = (const float*)d_in[4];
    const float* bk = (const float*)d_in[5];
    const float* Wv = (const float*)d_in[6];
    const float* bv = (const float*)d_in[7];
    const float* Wo = (const float*)d_in[8];
    const float* bo = (const float*)d_in[9];
    const float* Wg = (const float*)d_in[10];
    const float* bg = (const float*)d_in[11];
    const float* W1 = (const float*)d_in[12];
    const float* b1 = (const float*)d_in[13];
    const float* W2 = (const float*)d_in[14];
    const float* b2 = (const float*)d_in[15];
    const float* ln1_g = (const float*)d_in[16];
    const float* ln1_b = (const float*)d_in[17];
    const float* ln2_g = (const float*)d_in[18];
    const float* ln2_b = (const float*)d_in[19];
    const int* leaf = (const int*)d_in[21];   // mask (d_in[20]) derived from leaf on device

    float* x  = (float*)d_out;
    float* ws = (float*)d_ws;

    const size_t nBLH = (size_t)BB * LL * HH;     // 2M
    float* h    = ws;
    float* qb   = h + nBLH;
    float* kb2  = qb + nBLH;
    float* vb   = kb2 + nBLH;
    float* ctx  = vb + nBLH;
    float* ff   = ctx + nBLH;
    float* gate = ff + (size_t)BB * LL * DFFv;
    float* sinb = gate + (size_t)BB * LL * NHH;
    float* cosb = sinb + (size_t)BB * LL * (DKK / 2);
    float* pk0  = cosb + (size_t)BB * LL * (DKK / 2);
    __bf16*   qpk = (__bf16*)pk0;                        // 2M bf16 = 1M floats each
    __bf16*   qrt = (__bf16*)(pk0 + nBLH / 2);
    __bf16*   kpk = (__bf16*)(pk0 + nBLH);
    __bf16*   krt = (__bf16*)(pk0 + 3 * nBLH / 2);
    _Float16* vpk = (_Float16*)(pk0 + 2 * nBLH);

    hipMemcpyAsync(x, content, nBLH * sizeof(float), hipMemcpyDeviceToDevice, stream);

    sincos_kernel<<<(BB * LL * (DKK / 2) + 255) / 256, 256, 0, stream>>>(leaf, sinb, cosb);

    const float scale = 0.125f;  // DK^-0.5
    const int M = BB * LL;       // 2048
    dim3 blk(256);
    dim3 gP(HH / 64, M / 128);
    dim3 gA(LL / 16, NHH, BB);
    dim3 gF1(DFFv / 64, M / 128);
    dim3 gF2(HH / 64, M / 128);
    const int packBlocks = (BB * LL * HH / 8) / 256;   // 262144 threads / 256

    for (int i = 0; i < NLv; i++) {
        const size_t oHH = (size_t)i * HH * HH;
        const size_t oH  = (size_t)i * HH;
        ln_kernel<<<M, 256, 0, stream>>>(x, ln1_g + oH, ln1_b + oH, h);

        gemm_kernel<<<gP, blk, 0, stream>>>(h, Wq + oHH, bq + oH, nullptr, qb, M, HH, HH, scale, 0);
        gemm_kernel<<<gP, blk, 0, stream>>>(h, Wk + oHH, bk + oH, nullptr, kb2, M, HH, HH, 1.0f, 0);
        gemm_kernel<<<gP, blk, 0, stream>>>(h, Wv + oHH, bv + oH, nullptr, vb, M, HH, HH, 1.0f, 0);
        gate_kernel<<<M, 128, 0, stream>>>(h, Wg + (size_t)i * HH * NHH, bg + (size_t)i * NHH, gate);

        pack_qk_kernel<<<packBlocks, 256, 0, stream>>>(qb, sinb, cosb, qpk, qrt);
        pack_qk_kernel<<<packBlocks, 256, 0, stream>>>(kb2, sinb, cosb, kpk, krt);
        pack_v_kernel<<<packBlocks, 256, 0, stream>>>(vb, vpk);

        attn_kernel<<<gA, blk, 0, stream>>>(qpk, qrt, kpk, krt, vpk, gate, tree_score, leaf, ctx);

        gemm_kernel<<<gP, blk, 0, stream>>>(ctx, Wo + oHH, bo + oH, x, x, M, HH, HH, 1.0f, 0);

        ln_kernel<<<M, 256, 0, stream>>>(x, ln2_g + oH, ln2_b + oH, h);

        gemm_kernel<<<gF1, blk, 0, stream>>>(h, W1 + (size_t)i * HH * DFFv, b1 + (size_t)i * DFFv,
                                             nullptr, ff, M, DFFv, HH, 1.0f, 1);
        gemm_kernel<<<gF2, blk, 0, stream>>>(ff, W2 + (size_t)i * DFFv * HH, b2 + oH,
                                             x, x, M, HH, DFFv, 1.0f, 0);
    }
}
